// TmsFastAutoencoder_30983894073784
// MI455X (gfx1250) — matrix operand order repaired and checked by reading, not compiled, not run on hardware
//
#include <hip/hip_runtime.h>
#include <math.h>

// ---------------- problem constants (fixed by the reference) ----------------
#define BDIM 2048
#define DDIM 2048
#define FDIM 65536
#define KTOP 64
#define AUXK 256
#define DEAD_THRESH 5
#define NZ_EPS 0.001f

// ---------------- CDNA5 vector types ----------------
typedef __attribute__((ext_vector_type(16))) __bf16 bf16x16;
typedef __attribute__((ext_vector_type(8)))  __bf16 bf16x8;
typedef __attribute__((ext_vector_type(4)))  __bf16 bf16x4;
typedef __attribute__((ext_vector_type(8)))  float  f32x8;
typedef __attribute__((ext_vector_type(4)))  int    i32x4;

// Async global->LDS path (gfx1250), guarded so either toolchain compiles.
// Probe (round 2 diagnostic) showed arg0 of the builtin is an int4* in the
// global (__device__ / AS1) address space.
#if __has_builtin(__builtin_amdgcn_global_load_async_to_lds_b128)
#define HAVE_ASYNC 1
typedef __attribute__((address_space(1))) i32x4* gv4_ptr;
typedef __attribute__((address_space(3))) i32x4* lv4_ptr;
#endif

__device__ __forceinline__ void async_fence() {
#ifdef HAVE_ASYNC
#if __has_builtin(__builtin_amdgcn_s_wait_asynccnt)
  __builtin_amdgcn_s_wait_asynccnt(0);
#else
  asm volatile("s_wait_asynccnt 0" ::: "memory");
#endif
#endif
}

// ---------------- kernel 1: xb = bf16(x - pre_bias) ----------------
__global__ void prep_x_kernel(const float* __restrict__ x,
                              const float* __restrict__ pre_bias,
                              __bf16* __restrict__ xb) {
  int i = blockIdx.x * 256 + threadIdx.x;          // over B*D = 4M
  int d = i & (DDIM - 1);
  xb[i] = (__bf16)(x[i] - pre_bias[d]);
}

// ------- kernel 2: Wt[f][d] = bf16(W_enc[d][f])  (tiled LDS transpose) ------
#define TT 64
__global__ __launch_bounds__(256)
void conv_wT_kernel(const float* __restrict__ W, __bf16* __restrict__ Wt) {
  __shared__ __bf16 tile[TT][TT + 8];              // +16B pad
  const int f0 = blockIdx.x * TT;                  // along F
  const int d0 = blockIdx.y * TT;                  // along D
  const int t  = threadIdx.x;
  // read coalesced along F, store transposed into LDS
#pragma unroll
  for (int i = 0; i < (TT * TT) / 256; ++i) {
    int j = t + i * 256;
    int r = j >> 6;                                // d offset
    int c = j & 63;                                // f offset
    tile[c][r] = (__bf16)W[(size_t)(d0 + r) * FDIM + f0 + c];
  }
  __syncthreads();
  // write coalesced along D, 16B vector stores
#pragma unroll
  for (int i = 0; i < (TT * TT) / (8 * 256); ++i) {
    int j = t + i * 256;                           // chunk id, 512 chunks
    int r = j >> 3;                                // f offset
    int c8 = j & 7;                                // d chunk
    *(bf16x8*)&Wt[(size_t)(f0 + r) * DDIM + d0 + c8 * 8] =
        *(const bf16x8*)&tile[r][c8 * 8];
  }
}

// ---------------- kernel 3: bf16 WMMA GEMM, latents = xb @ Wt^T -------------
// Block tile BM=128 x BN=256, BK=32. 256 threads = 8 wave32 waves in a
// 2(M) x 4(N) grid; each wave computes a 64x64 = 4x4 set of 16x16 WMMA tiles.
#define BM 128
#define BN 256
#define BK 32
#define LDP 40   // padded LDS row stride (bf16): 80B keeps 16B-aligned chunks

union AFrag { bf16x16 v; bf16x8 h[2]; };

__global__ __launch_bounds__(256)
void gemm_bf16_kernel(const __bf16* __restrict__ xb,
                      const __bf16* __restrict__ Wt,   // [F][D]
                      float* __restrict__ lat) {
  __shared__ __bf16 As[2][BM][LDP];   // [m][k]
  __shared__ __bf16 Bs[2][BN][LDP];   // [n][k]

  const int tid  = threadIdx.x;
  const int lane = tid & 31;
  const int wave = tid >> 5;
  const int wm   = wave & 1;          // M offset wm*64
  const int wn   = wave >> 1;         // N offset wn*64
  const int lh   = lane >> 4;
  const int lm   = lane & 15;

  const int row0 = blockIdx.x * BM;   // M fast -> W N-panel stays in L2
  const int col0 = blockIdx.y * BN;

  f32x8 acc[4][4] = {};

  auto stage = [&](int buf, int kk) {
#pragma unroll
    for (int i = 0; i < (BM * BK) / (8 * 256); ++i) {   // 2 chunks of 16B
      int j = tid + i * 256;
      int r = j >> 2, c8 = j & 3;
      const __bf16* g = &xb[(size_t)(row0 + r) * DDIM + kk + c8 * 8];
      __bf16* l = &As[buf][r][c8 * 8];
#ifdef HAVE_ASYNC
      __builtin_amdgcn_global_load_async_to_lds_b128((gv4_ptr)g, (lv4_ptr)l, 0, 0);
#else
      *(bf16x8*)l = *(const bf16x8*)g;
#endif
    }
#pragma unroll
    for (int i = 0; i < (BN * BK) / (8 * 256); ++i) {   // 4 chunks of 16B
      int j = tid + i * 256;
      int n = j >> 2, k8 = j & 3;
      const __bf16* g = &Wt[(size_t)(col0 + n) * DDIM + kk + k8 * 8];
      __bf16* l = &Bs[buf][n][k8 * 8];
#ifdef HAVE_ASYNC
      __builtin_amdgcn_global_load_async_to_lds_b128((gv4_ptr)g, (lv4_ptr)l, 0, 0);
#else
      *(bf16x8*)l = *(const bf16x8*)g;
#endif
    }
  };

  stage(0, 0);
  async_fence();
  __syncthreads();

  int cur = 0;
  for (int k0 = 0; k0 < DDIM; k0 += BK) {
    if (k0 + BK < DDIM) {
      stage(cur ^ 1, k0 + BK);                     // overlap with compute
#ifndef HAVE_ASYNC
      __builtin_prefetch(&xb[(size_t)(row0 + (tid >> 1)) * DDIM + k0 + BK], 0, 1);
#endif
    }

    // A fragments: lane<16 -> K 0..15; lane>=16 -> K 16..31
    AFrag a[4];
#pragma unroll
    for (int mt = 0; mt < 4; ++mt) {
      int r = wm * 64 + mt * 16 + lm;
      int klo = lh * 16;
      a[mt].h[0] = *(const bf16x8*)&As[cur][r][klo];
      a[mt].h[1] = *(const bf16x8*)&As[cur][r][klo + 8];
    }
    // B fragments: lane<16 -> K 0..15 of its column, lane>=16 -> K 16..31
    bf16x16 b[4];
#pragma unroll
    for (int nt = 0; nt < 4; ++nt) {
      int c = wn * 64 + nt * 16 + lm;
      b[nt] = *(const bf16x16*)&Bs[cur][c][lh * 16];
    }
#pragma unroll
    for (int mt = 0; mt < 4; ++mt)
#pragma unroll
      for (int nt = 0; nt < 4; ++nt)
        acc[mt][nt] = __builtin_amdgcn_wmma_f32_16x16x32_bf16(
            false, a[mt].v, false, b[nt], (short)0, acc[mt][nt], false, false);

    async_fence();
    __syncthreads();
    cur ^= 1;
  }

  // epilogue: C layout -> VGPR v, lane l: M = v + 8*(l>=16), N = l&15
#pragma unroll
  for (int mt = 0; mt < 4; ++mt)
#pragma unroll
    for (int nt = 0; nt < 4; ++nt) {
      int rb = row0 + wm * 64 + mt * 16 + (lh ? 8 : 0);
      int cb = col0 + wn * 64 + nt * 16 + lm;
#pragma unroll
      for (int v = 0; v < 8; ++v)
        lat[(size_t)(rb + v) * FDIM + cb] = acc[mt][nt][v];
    }
}

// ---------------- kernel 4/6: per-row top-R by cached argmax extraction -----
__global__ __launch_bounds__(256)
void row_topk_kernel(const float* __restrict__ lat,
                     const float* __restrict__ bias,   // nullable
                     const float* __restrict__ mask,   // nullable
                     int rounds,
                     int*   __restrict__ out_idx,
                     float* __restrict__ out_val,
                     int*   __restrict__ fired) {      // nullable
  const int b = blockIdx.x, t = threadIdx.x;
  __shared__ float rv[256];
  __shared__ int   rf[256];
  __shared__ int   rt[256];

  float v[FDIM / 256];
  const float* row = lat + (size_t)b * FDIM;
  float mv = -3.0e38f; int mi = 0;
  for (int i = 0; i < FDIM / 256; ++i) {
    int f = i * 256 + t;                  // coalesced
    float x = row[f];
    if (bias) x += bias[f];
    if (mask) x *= mask[f];
    v[i] = x;
    if (x > mv) { mv = x; mi = i; }
  }

  for (int r = 0; r < rounds; ++r) {
    rv[t] = mv; rf[t] = mi * 256 + t; rt[t] = t;
    __syncthreads();
    for (int off = 128; off > 0; off >>= 1) {
      if (t < off && rv[t + off] > rv[t]) {
        rv[t] = rv[t + off]; rf[t] = rf[t + off]; rt[t] = rt[t + off];
      }
      __syncthreads();
    }
    float wv = rv[0]; int wf = rf[0]; int wt = rt[0];
    __syncthreads();
    if (t == 0) {
      float rel = wv > 0.f ? wv : 0.f;
      out_idx[(size_t)b * rounds + r] = wf;
      out_val[(size_t)b * rounds + r] = rel;
      if (fired && rel > NZ_EPS) atomicOr(&fired[wf], 1);
    }
    if (t == wt) {                        // winner pops & rescans locally
      v[mi] = -3.0e38f;
      mv = -3.0e38f; mi = 0;
      for (int i = 0; i < FDIM / 256; ++i)
        if (v[i] > mv) { mv = v[i]; mi = i; }
    }
    __syncthreads();
  }
}

// ---------------- kernel 5a: zero the fired flags ----------------
__global__ void zero_fired_kernel(int* __restrict__ fired) {
  fired[blockIdx.x * 256 + threadIdx.x] = 0;
}

// ---------------- kernel 5b: dead mask from stats + fired ----------------
__global__ void dead_mask_kernel(const long long* __restrict__ stats_old,
                                 const int* __restrict__ fired,
                                 float* __restrict__ dead) {
  int f = blockIdx.x * 256 + threadIdx.x;
  long long s = fired[f] ? 1LL : (stats_old[f] + 1LL);
  dead[f] = (s > (long long)DEAD_THRESH) ? 1.0f : 0.0f;
}

// ---------------- launch ----------------
extern "C" void kernel_launch(void* const* d_in, const int* in_sizes, int n_in,
                              void* d_out, int out_size, void* d_ws, size_t ws_size,
                              hipStream_t stream) {
  const float*     x         = (const float*)d_in[0];
  const float*     pre_bias  = (const float*)d_in[1];
  const float*     W_enc     = (const float*)d_in[2];
  const float*     latent_b  = (const float*)d_in[3];
  const long long* stats_old = (const long long*)d_in[4];
  // d_in[5..7] = k, auxk, dead_steps_threshold (fixed; hardcoded above)

  char* w = (char*)d_ws;
  __bf16* xb = (__bf16*)w;                                   // 8 MB
  w += (size_t)BDIM * DDIM * sizeof(__bf16);
  __bf16* Wt = (__bf16*)w;                                   // 256 MB, [F][D]
  w += (size_t)DDIM * FDIM * sizeof(__bf16);
  float* lat = (float*)w;                                    // 512 MB
  w += (size_t)BDIM * FDIM * sizeof(float);
  int* fired = (int*)w;                                      // 256 KB
  w += (size_t)FDIM * sizeof(int);
  float* dead = (float*)w;                                   // 256 KB

  // output layout: topk_idxs | topk_vals | auxk_idxs | auxk_vals
  int*   topk_idx = (int*)d_out;
  float* topk_val = (float*)d_out + (size_t)BDIM * KTOP;
  int*   auxk_idx = (int*)d_out + (size_t)2 * BDIM * KTOP;
  float* auxk_val = (float*)d_out + (size_t)2 * BDIM * KTOP + (size_t)BDIM * AUXK;

  prep_x_kernel<<<(BDIM * DDIM) / 256, 256, 0, stream>>>(x, pre_bias, xb);
  conv_wT_kernel<<<dim3(FDIM / TT, DDIM / TT), 256, 0, stream>>>(W_enc, Wt);
  gemm_bf16_kernel<<<dim3(BDIM / BM, FDIM / BN), 256, 0, stream>>>(xb, Wt, lat);
  zero_fired_kernel<<<FDIM / 256, 256, 0, stream>>>(fired);
  row_topk_kernel<<<BDIM, 256, 0, stream>>>(lat, latent_b, nullptr, KTOP,
                                            topk_idx, topk_val, fired);
  dead_mask_kernel<<<FDIM / 256, 256, 0, stream>>>(stats_old, fired, dead);
  row_topk_kernel<<<BDIM, 256, 0, stream>>>(lat, nullptr, dead, AUXK,
                                            auxk_idx, auxk_val, nullptr);
}
